// TransitionLayer_6811818131657
// MI455X (gfx1250) — matrix operand order repaired and checked by reading, not compiled, not run on hardware
//
#include <hip/hip_runtime.h>
#include <hip/hip_bf16.h>

#define CN 8192   // code_num
#define GD 512    // graph_size
#define HD 512    // hidden
#define AD 128    // attn dim
#define OD 512    // output dim

typedef __attribute__((ext_vector_type(16))) _Float16     v16h;
typedef __attribute__((ext_vector_type(8)))  float        v8f;
typedef __attribute__((ext_vector_type(4)))  unsigned int v4u;
typedef __attribute__((ext_vector_type(8)))  int          v8i;
typedef __attribute__((ext_vector_type(4)))  int          v4i;

// ---------------------------------------------------------------------------
// WMMA helpers (CDNA5 wave32, 16x16x32 f16 -> f32)
// ---------------------------------------------------------------------------
static __device__ inline v8f wmma16(v16h a, v16h b, v8f c) {
  return __builtin_amdgcn_wmma_f32_16x16x32_f16(
      /*neg_a=*/false, a, /*neg_b=*/false, b,
      /*c_mod=*/(short)0, c, /*reuse_a=*/false, /*reuse_b=*/false);
}

// A fragment: LDS row-major [M][K]. ISA 16-bit A 16x32 layout: lane holds row
// M=lane&15; elems 0..7 -> K=koff..koff+7, elems 8..15 -> K=koff+16..koff+23,
// koff = (lane>=16)*8.
static __device__ inline v16h frag_a(const _Float16* s, int ld, int mb, int kb) {
  const int lane = threadIdx.x & 31;
  const _Float16* p = s + (size_t)(mb + (lane & 15)) * ld + kb + ((lane >> 4) << 3);
  v16h f;
#pragma unroll
  for (int e = 0; e < 8; ++e) { f[e] = p[e]; f[e + 8] = p[e + 16]; }
  return f;
}

// B fragment: LDS [N][K] (K contiguous). Lane holds column N=lane&15;
// elems 0..7 -> K=koff..koff+7, elems 8..15 -> K=koff+16..koff+23, koff=(lane>=16)*8.
static __device__ inline v16h frag_b(const _Float16* s, int ld, int nb, int kb) {
  const int lane = threadIdx.x & 31;
  const _Float16* p = s + (size_t)(nb + (lane & 15)) * ld + kb + ((lane >> 4) << 3);
  v16h f;
#pragma unroll
  for (int e = 0; e < 8; ++e) { f[e] = p[e]; f[e + 8] = p[e + 16]; }
  return f;
}

// Float atomic max via monotonic IEEE int encoding (init target to -inf).
static __device__ inline void atomic_max_f32(float* a, float v) {
  if (v >= 0.f) atomicMax((int*)a, __float_as_int(v));
  else          atomicMin((unsigned int*)a, __float_as_uint(v));
}

// ---------------------------------------------------------------------------
// Tensor Data Mover: DMA a tile from global into LDS (ASIC copies, no VGPRs).
// data_size = 8 bytes/unit; td0 = row length in 8B units; td1 = #rows;
// stride0 = global row stride in 8B units. Tiles are constructed in-bounds,
// so tensor dims are set generously (no OOB zero-fill needed).
// D# packing per CDNA5 ISA sec 8.3/8.4 (group0: count/lds/global/type,
// group1: data_size/tensor dims/tile dims/stride).
// This toolchain exposes the 6-arg builtin (g0, g1, g2, g3, g4, cpol).
// ---------------------------------------------------------------------------
static __device__ inline void tdm_load_tile(unsigned lds, const void* g,
                                            unsigned td0, unsigned td1,
                                            unsigned stride0) {
  const unsigned long long ga = (unsigned long long)(uintptr_t)g;
  const unsigned tensor_d0 = 0x7FFFFFFFu;
  const unsigned tensor_d1 = 0xFFFFu;
  v4u g0;
  g0[0] = 1u;                                   // count=1 (valid), load, user
  g0[1] = lds;                                  // lds_addr (bytes)
  g0[2] = (unsigned)ga;                         // global_addr[31:0]
  g0[3] = (unsigned)((ga >> 32) & 0x01FFFFFFu)  // global_addr[56:32]
          | 0x80000000u;                        // type=2 ("image")
  v8i g1;
  g1[0] = (int)(3u << 16);                      // data_size=3 (8B), mask=0
  g1[1] = (int)((tensor_d0 & 0xFFFFu) << 16);   // tensor_dim0[15:0]
  g1[2] = (int)((tensor_d0 >> 16) | ((tensor_d1 & 0xFFFFu) << 16));
  g1[3] = (int)((tensor_d1 >> 16) | (td0 << 16));  // tile_dim0
  g1[4] = (int)(td1 & 0xFFFFu);                 // tile_dim1 (tile_dim2=0)
  g1[5] = (int)stride0;                         // tensor_dim0_stride[31:0]
  g1[6] = 0;
  g1[7] = 0;
  const v4i z4 = {0, 0, 0, 0};
  const v8i z8 = {0, 0, 0, 0, 0, 0, 0, 0};
  __builtin_amdgcn_tensor_load_to_lds(g0, g1, z4, z4, z8, 0);
}

template <typename T>
static __device__ inline unsigned lds_off(T* p) {
  return (unsigned)(unsigned long long)(uintptr_t)p;  // LDS aperture: addr[31:0]
}

// ---------------------------------------------------------------------------
// Init: zero h_new, -inf output vector, zero compaction counters
// ---------------------------------------------------------------------------
__global__ __launch_bounds__(256) void init_kernel(float* outvec, float* h_new,
                                                   int* counters) {
  const int i = blockIdx.x * 256 + threadIdx.x;
  if (i < CN * HD) h_new[i] = 0.f;
  if (i < OD)      outvec[i] = -__builtin_inff();
  if (i < 2)       counters[i] = 0;
}

// ---------------------------------------------------------------------------
// Mask compaction: idx1 = rows with m1, idx23 = rows with m2|m3 (+sel: m3)
// ---------------------------------------------------------------------------
__global__ __launch_bounds__(256) void compact_kernel(
    const int* __restrict__ divided, int* counters,
    int* idx1, int* idx23, int* sel23) {
  const int i = blockIdx.x * 256 + threadIdx.x;
  if (i >= CN) return;
  const int d0 = divided[3 * i], d1 = divided[3 * i + 1], d2 = divided[3 * i + 2];
  if (d0 > 0) idx1[atomicAdd(&counters[0], 1)] = i;
  if (d1 > 0 || d2 > 0) {
    const int p = atomicAdd(&counters[1], 1);
    idx23[p] = i;
    sel23[p] = (d2 > 0) ? 1 : 0;
  }
}

// ---------------------------------------------------------------------------
// Gather-GEMM:  out[r, n] = sum_k Asel[rowidx[r]][k] * W[n][k] + bias[n]
// f32 sources converted to f16 while staging to LDS; f32 WMMA accumulate.
// Block: 256 threads (8 waves), tile 128x128, K-step 32. Wave tile 32x64.
// trans!=0: store f16 output transposed with leading dim ldT (V producer).
// ---------------------------------------------------------------------------
#define BM 128
#define BN 128
#define BK 32

__global__ __launch_bounds__(256) void gemm_kernel(
    const float* __restrict__ A0, const float* __restrict__ A1,
    const int* __restrict__ rowidx, const int* __restrict__ sel,
    const float* __restrict__ W, const float* __restrict__ bias,
    float* __restrict__ outF32, _Float16* __restrict__ outF16,
    int N, int K, int trans, int ldT, const int* __restrict__ mcnt) {
  const int M = *mcnt;
  const int m0 = blockIdx.y * BM;
  if (m0 >= M) return;
  const int n0 = blockIdx.x * BN;

  __shared__ _Float16 As[BM][BK];
  __shared__ _Float16 Bs[BN][BK];

  const int tid  = threadIdx.x;
  const int lane = tid & 31;
  const int wv   = tid >> 5;
  const int wm   = (wv >> 1) << 5;  // wave M offset: 0,32,64,96
  const int wn   = (wv & 1) << 6;   // wave N offset: 0,64

  const v8f zero = {};
  v8f acc[2][4];
#pragma unroll
  for (int a = 0; a < 2; ++a)
#pragma unroll
    for (int b = 0; b < 4; ++b) acc[a][b] = zero;

  for (int kk = 0; kk < K; kk += BK) {
    __syncthreads();
    // Stage A tile (gathered rows, f32 -> f16) + prefetch next K-step
    for (int u = tid; u < BM * (BK / 4); u += 256) {
      const int m = u >> 3, kc = (u & 7) << 2;
      const int gm = m0 + m;
      float4 f = make_float4(0.f, 0.f, 0.f, 0.f);
      if (gm < M) {
        const int row = rowidx[gm];
        const float* Ap = (sel && sel[gm]) ? A1 : A0;
        f = *reinterpret_cast<const float4*>(Ap + (size_t)row * K + kk + kc);
        if (kk + BK < K)
          __builtin_prefetch(Ap + (size_t)row * K + kk + BK + kc, 0, 0);
      }
      As[m][kc + 0] = (_Float16)f.x; As[m][kc + 1] = (_Float16)f.y;
      As[m][kc + 2] = (_Float16)f.z; As[m][kc + 3] = (_Float16)f.w;
    }
    // Stage B tile: Bs[n][k] = W[n0+n][kk+k]
    for (int u = tid; u < BN * (BK / 4); u += 256) {
      const int n = u >> 3, kc = (u & 7) << 2;
      const int gn = n0 + n;
      float4 f = make_float4(0.f, 0.f, 0.f, 0.f);
      if (gn < N) {
        f = *reinterpret_cast<const float4*>(W + (size_t)gn * K + kk + kc);
        if (kk + BK < K)
          __builtin_prefetch(W + (size_t)gn * K + kk + BK + kc, 0, 0);
      }
      Bs[n][kc + 0] = (_Float16)f.x; Bs[n][kc + 1] = (_Float16)f.y;
      Bs[n][kc + 2] = (_Float16)f.z; Bs[n][kc + 3] = (_Float16)f.w;
    }
    __syncthreads();

    const v16h a0 = frag_a(&As[0][0], BK, wm, 0);
    const v16h a1 = frag_a(&As[0][0], BK, wm + 16, 0);
#pragma unroll
    for (int nt = 0; nt < 4; ++nt) {
      const v16h b = frag_b(&Bs[0][0], BK, wn + (nt << 4), 0);
      acc[0][nt] = wmma16(a0, b, acc[0][nt]);
      acc[1][nt] = wmma16(a1, b, acc[1][nt]);
    }
  }

  // Epilogue: bias + store. C/D layout: lane = N&15, row = r + 8*(lane>=16)
  const int roff = (lane >> 4) << 3;
#pragma unroll
  for (int mt = 0; mt < 2; ++mt)
#pragma unroll
    for (int nt = 0; nt < 4; ++nt) {
      const int gn = n0 + wn + (nt << 4) + (lane & 15);
      if (gn >= N) continue;
      const float bb = bias[gn];
#pragma unroll
      for (int r = 0; r < 8; ++r) {
        const int gm = m0 + wm + (mt << 4) + r + roff;
        if (gm >= M) continue;
        const float v = acc[mt][nt][r] + bb;
        if (outF32)      outF32[(size_t)gm * N + gn] = v;
        else if (!trans) outF16[(size_t)gm * N + gn] = (_Float16)v;
        else             outF16[(size_t)gn * ldT + gm] = (_Float16)v;
      }
    }
}

// ---------------------------------------------------------------------------
// GRU elementwise over compacted m1 rows + atomic max into output vector
// ---------------------------------------------------------------------------
__global__ __launch_bounds__(256) void gru_kernel(
    const float* __restrict__ gi, const float* __restrict__ gh,
    const float* __restrict__ h0, const int* __restrict__ idx1,
    const int* __restrict__ mcnt, float* __restrict__ h_new,
    float* __restrict__ outvec) {
  const int i = blockIdx.x * 256 + threadIdx.x;
  const int r = i >> 9;           // / HD
  const int h = i & (HD - 1);
  if (r >= *mcnt) return;
  const int g = idx1[r];
  const size_t b = (size_t)r * (3 * HD);
  const float ir = gi[b + h], iz = gi[b + HD + h], in_ = gi[b + 2 * HD + h];
  const float hr = gh[b + h], hz = gh[b + HD + h], hn  = gh[b + 2 * HD + h];
  const float rg = 1.f / (1.f + expf(-(ir + hr)));
  const float z  = 1.f / (1.f + expf(-(iz + hz)));
  const float nn = tanhf(in_ + rg * hn);
  const float hg = (1.f - z) * nn + z * h0[(size_t)g * HD + h];
  h_new[(size_t)g * HD + h] = hg;
  atomic_max_f32(&outvec[h], hg);
}

// ---------------------------------------------------------------------------
// Flash attention over compacted m23 rows/cols.
// Block = 16 Q-rows x full O=512; 8 waves each own a 64-wide O-chunk.
// K/Q/V tiles are DMA'd into LDS by the Tensor Data Mover (wave 0 issues,
// s_wait_tensorcnt, then workgroup barrier releases consumers).
// V arrives pre-transposed (VcT[o][row]) so its tile lands directly in the
// [o][j] layout the WMMA B-fragments read.
// ---------------------------------------------------------------------------
__global__ __launch_bounds__(256) void attn_kernel(
    const _Float16* __restrict__ Qc, const _Float16* __restrict__ Kc,
    const _Float16* __restrict__ VcT, const int* __restrict__ idx23,
    const int* __restrict__ mcnt, float* __restrict__ h_new,
    float* __restrict__ outvec) {
  const int M = *mcnt;
  const int m0 = blockIdx.x << 4;
  if (m0 >= M) return;

  __shared__ _Float16 Qs[16][AD];     // 4 KB
  __shared__ _Float16 Ks[32][AD];     // 8 KB
  __shared__ _Float16 Vst[OD][32];    // 32 KB ([o][j])
  __shared__ float    Ss[16][32];     // 2 KB
  __shared__ _Float16 Ps[16][32];     // 1 KB
  __shared__ float rowM[16], rowL[16], rowA[16];

  const int tid  = threadIdx.x;
  const int lane = tid & 31;
  const int wv   = tid >> 5;
  const int och  = wv << 6;           // wave's 64-wide O chunk
  const float scale = 0.08838834764831845f;  // 1/sqrt(128)

  // Q tile: one contiguous 4KB DMA
  if (wv == 0) {
    tdm_load_tile(lds_off(&Qs[0][0]), Qc + (size_t)m0 * AD,
                  /*td0=*/16 * AD * 2 / 8, /*td1=*/1, /*stride0=*/16 * AD * 2 / 8);
    __builtin_amdgcn_s_wait_tensorcnt(0);
  }
  if (tid < 16) { rowM[tid] = -__builtin_inff(); rowL[tid] = 0.f; rowA[tid] = 0.f; }

  const v8f zero = {};
  v8f acc[4];
#pragma unroll
  for (int nt = 0; nt < 4; ++nt) acc[nt] = zero;

  for (int jb = 0; jb < M; jb += 32) {
    __syncthreads();   // previous iteration's consumers done with Ks/Vst/Ps
    if (wv == 0) {
      // K block: contiguous 8KB (32 rows x 256B)
      tdm_load_tile(lds_off(&Ks[0][0]), Kc + (size_t)jb * AD,
                    /*td0=*/32 * AD * 2 / 8, /*td1=*/1, /*stride0=*/32 * AD * 2 / 8);
      // V block: 512 rows x 64B, global row stride = CN f16
      tdm_load_tile(lds_off(&Vst[0][0]), VcT + jb,
                    /*td0=*/32 * 2 / 8, /*td1=*/OD, /*stride0=*/CN * 2 / 8);
      __builtin_amdgcn_s_wait_tensorcnt(0);
    }
    __syncthreads();   // LDS K/V ready for all waves

    // S = Q @ K^T (16x32), split over waves 0 and 1
    if (wv < 2) {
      v8f s = zero;
#pragma unroll
      for (int kc = 0; kc < AD; kc += 32)
        s = wmma16(frag_a(&Qs[0][0], AD, 0, kc),
                   frag_b(&Ks[0][0], AD, wv << 4, kc), s);
      const int n = (wv << 4) + (lane & 15);
      const int ro = (lane >> 4) << 3;
#pragma unroll
      for (int r = 0; r < 8; ++r) Ss[r + ro][n] = s[r] * scale;
    }
    __syncthreads();

    // Online softmax bookkeeping (one thread per row); cols >= M masked out
    if (tid < 16) {
      int lim = M - jb; if (lim > 32) lim = 32;
      float bm = -__builtin_inff();
      for (int c = 0; c < lim; ++c) bm = fmaxf(bm, Ss[tid][c]);
      const float nm = fmaxf(rowM[tid], bm);
      const float al = expf(rowM[tid] - nm);
      float l = rowL[tid] * al;
      for (int c = 0; c < 32; ++c) {
        const float p = (c < lim) ? expf(Ss[tid][c] - nm) : 0.f;
        Ps[tid][c] = (_Float16)p;
        l += p;
      }
      rowM[tid] = nm; rowL[tid] = l; rowA[tid] = al;
    }
    __syncthreads();

    // Rescale accumulators and add P @ V_block
    const int ro = (lane >> 4) << 3;
    float al8[8];
#pragma unroll
    for (int r = 0; r < 8; ++r) al8[r] = rowA[r + ro];
    const v16h pa = frag_a(&Ps[0][0], 32, 0, 0);
#pragma unroll
    for (int nt = 0; nt < 4; ++nt) {
#pragma unroll
      for (int r = 0; r < 8; ++r) acc[nt][r] *= al8[r];
      acc[nt] = wmma16(pa, frag_b(&Vst[0][0], 32, och + (nt << 4), 0), acc[nt]);
    }
  }

  // Epilogue: normalize, tanh, scatter h_new rows, block-max into output vector
  const int ro = (lane >> 4) << 3;
  float li[8];
#pragma unroll
  for (int r = 0; r < 8; ++r) li[r] = rowL[r + ro];
#pragma unroll
  for (int nt = 0; nt < 4; ++nt) {
    const int n = och + (nt << 4) + (lane & 15);
    float vmax = -__builtin_inff();
#pragma unroll
    for (int r = 0; r < 8; ++r) {
      const int gr = m0 + r + ro;
      if (gr < M) {
        const float val = tanhf(acc[nt][r] / li[r]);
        h_new[(size_t)idx23[gr] * OD + n] = val;
        vmax = fmaxf(vmax, val);
      }
    }
    vmax = fmaxf(vmax, __shfl_xor(vmax, 16, 32));
    if (lane < 16) atomic_max_f32(&outvec[och + (nt << 4) + lane], vmax);
  }
}

// ---------------------------------------------------------------------------
// Host launch
// ---------------------------------------------------------------------------
static inline int cdiv(int a, int b) { return (a + b - 1) / b; }

extern "C" void kernel_launch(void* const* d_in, const int* in_sizes, int n_in,
                              void* d_out, int out_size, void* d_ws, size_t ws_size,
                              hipStream_t stream) {
  (void)in_sizes; (void)n_in; (void)out_size; (void)ws_size;

  const float* co      = (const float*)d_in[1];
  const int*   divided = (const int*)  d_in[2];
  const float* no_e    = (const float*)d_in[3];
  const float* unrel   = (const float*)d_in[4];
  const float* h0      = (const float*)d_in[5];
  const float* w_ih    = (const float*)d_in[6];
  const float* w_hh    = (const float*)d_in[7];
  const float* b_ih    = (const float*)d_in[8];
  const float* b_hh    = (const float*)d_in[9];
  const float* wq      = (const float*)d_in[10];
  const float* bq      = (const float*)d_in[11];
  const float* wk      = (const float*)d_in[12];
  const float* bk      = (const float*)d_in[13];
  const float* wv      = (const float*)d_in[14];
  const float* bv      = (const float*)d_in[15];

  float* outvec = (float*)d_out;        // [512]
  float* h_new  = outvec + OD;          // [C, 512]

  // Workspace carve-up
  char* w = (char*)d_ws;
  int* counters = (int*)w;              w += 256;
  int* idx1     = (int*)w;              w += CN * 4;
  int* idx23    = (int*)w;              w += CN * 4;
  int* sel23    = (int*)w;              w += CN * 4;
  float* gi     = (float*)w;            w += (size_t)CN * 3 * HD * 4;
  float* gh     = (float*)w;            w += (size_t)CN * 3 * HD * 4;
  _Float16* Qc  = (_Float16*)w;         w += (size_t)CN * AD * 2;
  _Float16* Kc  = (_Float16*)w;         w += (size_t)CN * AD * 2;
  _Float16* VcT = (_Float16*)w;         w += (size_t)CN * OD * 2;  // [OD][CN]

  const dim3 blk(256);

  init_kernel<<<cdiv(CN * HD, 256), blk, 0, stream>>>(outvec, h_new, counters);
  compact_kernel<<<cdiv(CN, 256), blk, 0, stream>>>(divided, counters, idx1, idx23, sel23);

  // GRU gate GEMMs over compacted m1 rows
  gemm_kernel<<<dim3((3 * HD) / BN, CN / BM), blk, 0, stream>>>(
      co, nullptr, idx1, nullptr, w_ih, b_ih, gi, nullptr, 3 * HD, GD, 0, 0, counters + 0);
  gemm_kernel<<<dim3((3 * HD) / BN, CN / BM), blk, 0, stream>>>(
      h0, nullptr, idx1, nullptr, w_hh, b_hh, gh, nullptr, 3 * HD, HD, 0, 0, counters + 0);
  gru_kernel<<<cdiv(CN * HD, 256), blk, 0, stream>>>(
      gi, gh, h0, idx1, counters + 0, h_new, outvec);

  // Q/K/V projections over compacted m23 rows (Q/K select no vs unrelated per row)
  gemm_kernel<<<dim3(1, CN / BM), blk, 0, stream>>>(
      no_e, unrel, idx23, sel23, wq, bq, nullptr, Qc, AD, GD, 0, 0, counters + 1);
  gemm_kernel<<<dim3(1, CN / BM), blk, 0, stream>>>(
      no_e, unrel, idx23, sel23, wk, bk, nullptr, Kc, AD, GD, 0, 0, counters + 1);
  gemm_kernel<<<dim3(OD / BN, CN / BM), blk, 0, stream>>>(
      co, nullptr, idx23, nullptr, wv, bv, nullptr, VcT, OD, GD, 1, CN, counters + 1);

  // Fused flash attention (TDM-staged tiles) + tanh + scatter + output max
  attn_kernel<<<CN / 16, blk, 0, stream>>>(
      Qc, Kc, VcT, idx23, counters + 1, h_new, outvec);
}
